// GALA_7962869367118
// MI455X (gfx1250) — matrix operand order repaired and checked by reading, not compiled, not run on hardware
//
#include <hip/hip_runtime.h>
#include <hip/hip_bf16.h>

typedef __attribute__((ext_vector_type(16))) _Float16 v16h;
typedef __attribute__((ext_vector_type(8)))  _Float16 v8h;
typedef __attribute__((ext_vector_type(8)))  float    v8f;
typedef __attribute__((ext_vector_type(4)))  float    v4f;

#define BB 8
#define NN 1024
#define CC 768
#define HH 12
#define HD 64
#define TC (3 * CC)

// ---------------------------------------------------------------- helpers

__device__ __forceinline__ v8f wmma_f16(v16h a, v16h b, v8f c) {
  // (neg_a, A, neg_b, B, c_mod, C, reuse_a, reuse_b)
  return __builtin_amdgcn_wmma_f32_16x16x32_f16(false, a, false, b, (short)0, c,
                                                false, false);
}

// A-matrix 16x32 f16 fragment: lanes 0-15/16-31 both hold rows 0-15,
// elements e<8 -> K = half*8 + e ; e>=8 -> K = 16 + half*8 + (e-8)
__device__ __forceinline__ v16h load_a_frag(const _Float16* __restrict__ row,
                                            int kb, int half) {
  const _Float16* p = row + kb + half * 8;
  v8h lo = *(const v8h*)p;
  v8h hi = *(const v8h*)(p + 16);
  v16h a;
#pragma unroll
  for (int i = 0; i < 8; ++i) {
    a[i] = lo[i];
    a[i + 8] = hi[i];
  }
  return a;
}

// B-matrix 32x16 f16 fragment from a row-major row (NT gemm: B[k][n]=Bt[n][k]):
// elements e<8 -> K = half*8 + e ; e>=8 -> K = 16 + half*8 + (e-8)
__device__ __forceinline__ v16h load_b_frag(const _Float16* __restrict__ row,
                                            int kb, int half) {
  const _Float16* p = row + kb + half * 8;
  v8h lo = *(const v8h*)p;
  v8h hi = *(const v8h*)(p + 16);
  v16h b;
#pragma unroll
  for (int i = 0; i < 8; ++i) {
    b[i] = lo[i];
    b[i + 8] = hi[i];
  }
  return b;
}

__device__ __forceinline__ float warp_sum(float v) {
#pragma unroll
  for (int o = 16; o > 0; o >>= 1) v += __shfl_xor(v, o, 32);
  return v;
}
__device__ __forceinline__ float warp_max(float v) {
#pragma unroll
  for (int o = 16; o > 0; o >>= 1) v = fmaxf(v, __shfl_xor(v, o, 32));
  return v;
}

__device__ __forceinline__ float block_sum(float v, float* red) {
  int lane = threadIdx.x & 31, wid = threadIdx.x >> 5;
  int nw = (blockDim.x + 31) >> 5;
  v = warp_sum(v);
  __syncthreads();
  if (lane == 0) red[wid] = v;
  __syncthreads();
  if (wid == 0) {
    float t = (lane < nw) ? red[lane] : 0.f;
    t = warp_sum(t);
    if (lane == 0) red[0] = t;
  }
  __syncthreads();
  return red[0];
}

__device__ __forceinline__ float block_max(float v, float* red) {
  int lane = threadIdx.x & 31, wid = threadIdx.x >> 5;
  int nw = (blockDim.x + 31) >> 5;
  v = warp_max(v);
  __syncthreads();
  if (lane == 0) red[wid] = v;
  __syncthreads();
  if (wid == 0) {
    float t = (lane < nw) ? red[lane] : -3.4e38f;
    t = warp_max(t);
    if (lane == 0) red[0] = t;
  }
  __syncthreads();
  return red[0];
}

// ---------------------------------------------------------------- kernels

__global__ void cvt_f32_to_f16(const float* __restrict__ in,
                               _Float16* __restrict__ out, int n) {
  int i = blockIdx.x * blockDim.x + threadIdx.x;
  if (i < n) out[i] = (_Float16)in[i];
}

__global__ void zero_f32(float* __restrict__ p, int n) {
  int i = blockIdx.x * blockDim.x + threadIdx.x;
  if (i < n) p[i] = 0.f;
}

// qkv = x @ w_in^T ; scatter f16 into q/k [B,H,N,HD] and v transposed
// vT [B,H,HD,N] so the PV gemm can use contiguous B-fragment loads.
__global__ void __launch_bounds__(32)
qkv_gemm(const _Float16* __restrict__ xh, const _Float16* __restrict__ wih,
         _Float16* __restrict__ qh, _Float16* __restrict__ kh,
         _Float16* __restrict__ vT) {
  const int lane = threadIdx.x;
  const int half = lane >> 4;
  const int l16 = lane & 15;
  const int rowbase = blockIdx.x * 16;   // token rows (B*N = 8192)
  const int colbase = blockIdx.y * 128;  // output cols (3C = 2304)

  const _Float16* arow = xh + (size_t)(rowbase + l16) * CC;
  const _Float16* brow[8];
#pragma unroll
  for (int dc = 0; dc < 8; ++dc)
    brow[dc] = wih + (size_t)(colbase + dc * 16 + l16) * CC;

  v8f acc[8] = {};
  for (int kb = 0; kb < CC; kb += 32) {
    v16h a = load_a_frag(arow, kb, half);
    __builtin_prefetch((const void*)(arow + kb + 32), 0, 3);
#pragma unroll
    for (int dc = 0; dc < 8; ++dc) {
      v16h bf = load_b_frag(brow[dc], kb, half);
      acc[dc] = wmma_f16(a, bf, acc[dc]);
    }
  }

#pragma unroll
  for (int dc = 0; dc < 8; ++dc) {
#pragma unroll
    for (int r = 0; r < 8; ++r) {
      int m = rowbase + r + 8 * half;
      int n = colbase + dc * 16 + l16;
      int sel = n / CC;
      int cc = n - sel * CC;
      int hh = cc >> 6;
      int dd = cc & 63;
      int bb = m >> 10;
      int tok = m & (NN - 1);
      _Float16 val = (_Float16)acc[dc][r];
      size_t head = (size_t)bb * HH + hh;
      if (sel == 0)
        qh[(head * NN + tok) * HD + dd] = val;
      else if (sel == 1)
        kh[(head * NN + tok) * HD + dd] = val;
      else
        vT[(head * HD + dd) * NN + tok] = val;  // transposed V
    }
  }
}

// One wave per (b, h, 16-query tile): QK^T -> softmax -> PV.
// Head-mean attention accumulated into aw via hardware f32 atomics.
__global__ void __launch_bounds__(32)
attn_kernel(const _Float16* __restrict__ qh, const _Float16* __restrict__ kh,
            const _Float16* __restrict__ vT, _Float16* __restrict__ ctxh,
            float* __restrict__ aw) {
  __shared__ float smp[16 * NN];  // 64KB: logits then probs for this tile

  const int lane = threadIdx.x;
  const int half = lane >> 4;
  const int l16 = lane & 15;
  const int qt = blockIdx.x;   // 0..63
  const int h = blockIdx.y;    // 0..11
  const int b = blockIdx.z;    // 0..7
  const float scale = 0.125f;  // 1/sqrt(64)
  const float invH = 1.0f / (float)HH;

  const size_t hbase = ((size_t)b * HH + h) * NN;
  float* awrow = aw + ((size_t)b * NN + qt * 16) * NN;  // 16 x 1024

  // Q fragments for 16 rows, hd=64 -> two k-chunks
  const _Float16* qrow = qh + (hbase + qt * 16 + l16) * HD;
  v16h a0 = load_a_frag(qrow, 0, half);
  v16h a1 = load_a_frag(qrow, 32, half);

  // logits: 64 key tiles of 16
  for (int kt = 0; kt < 64; ++kt) {
    const _Float16* krow = kh + (hbase + kt * 16 + l16) * HD;
    v16h b0 = load_b_frag(krow, 0, half);
    v16h b1 = load_b_frag(krow, 32, half);
    v8f c = {};
    c = wmma_f16(a0, b0, c);
    c = wmma_f16(a1, b1, c);
#pragma unroll
    for (int r = 0; r < 8; ++r)
      smp[(r + 8 * half) * NN + kt * 16 + l16] = c[r] * scale;
  }
  __syncthreads();

  // row softmax (all 32 lanes cooperate per row)
  for (int r = 0; r < 16; ++r) {
    float* row = smp + r * NN;
    float mx = -3.4e38f;
    for (int j = lane; j < NN; j += 32) mx = fmaxf(mx, row[j]);
    mx = warp_max(mx);
    float s = 0.f;
    for (int j = lane; j < NN; j += 32) {
      float e = __expf(row[j] - mx);
      row[j] = e;
      s += e;
    }
    s = warp_sum(s);
    float inv = 1.0f / s;
    float* aw_r = awrow + (size_t)r * NN;
    for (int j = lane; j < NN; j += 32) {
      float p = row[j] * inv;
      row[j] = p;
      unsafeAtomicAdd(aw_r + j, p * invH);  // global_atomic_add_f32
    }
  }
  __syncthreads();

  // ctx = P @ V   (16 x 64 output, 4 accumulators over d)
  v8f acc[4] = {};
  for (int kb = 0; kb < NN; kb += 32) {
    // A fragment from probs in LDS (row l16): four b128 DS loads + cvt
    const float* rowp = smp + l16 * NN + kb;
    v4f f0 = *(const v4f*)(rowp + half * 8);
    v4f f1 = *(const v4f*)(rowp + half * 8 + 4);
    v4f f2 = *(const v4f*)(rowp + 16 + half * 8);
    v4f f3 = *(const v4f*)(rowp + 16 + half * 8 + 4);
    v16h pa;
#pragma unroll
    for (int i = 0; i < 4; ++i) {
      pa[i] = (_Float16)f0[i];
      pa[i + 4] = (_Float16)f1[i];
      pa[i + 8] = (_Float16)f2[i];
      pa[i + 12] = (_Float16)f3[i];
    }
#pragma unroll
    for (int dc = 0; dc < 4; ++dc) {
      const _Float16* vrow = vT + (hbase * HD /*== head*HD*NN*/) +
                             (size_t)(dc * 16 + l16) * NN;
      v16h bf = load_b_frag(vrow, kb, half);
      acc[dc] = wmma_f16(pa, bf, acc[dc]);
    }
  }
#pragma unroll
  for (int dc = 0; dc < 4; ++dc) {
#pragma unroll
    for (int r = 0; r < 8; ++r) {
      int q = qt * 16 + r + 8 * half;
      int d = dc * 16 + l16;
      ctxh[((size_t)b * NN + q) * CC + h * HD + d] = (_Float16)acc[dc][r];
    }
  }
}

// importance[b,col] = sum_q |0.5*(aw[b,q,col+1]-aw[b,q,col-1])| (replicate edge)
__global__ void __launch_bounds__(256)
grad_kernel(const float* __restrict__ aw, float* __restrict__ imp) {
  int gid = blockIdx.x * blockDim.x + threadIdx.x;
  if (gid >= BB * NN) return;
  int b = gid >> 10;
  int col = gid & (NN - 1);
  int cm = (col > 0) ? col - 1 : 0;
  int cp = (col < NN - 1) ? col + 1 : NN - 1;
  const float* base = aw + (size_t)b * NN * NN;
  float s = 0.f;
  for (int q = 0; q < NN; ++q) {
    const float* row = base + (size_t)q * NN;
    s += fabsf(0.5f * (row[cp] - row[cm]));
  }
  imp[gid] = s;
}

// x_out = ctx @ w_out^T  (f32 output)
__global__ void __launch_bounds__(32)
outproj_gemm(const _Float16* __restrict__ ctxh, const _Float16* __restrict__ woh,
             float* __restrict__ xout) {
  const int lane = threadIdx.x;
  const int half = lane >> 4;
  const int l16 = lane & 15;
  const int rowbase = blockIdx.x * 16;
  const int colbase = blockIdx.y * 128;

  const _Float16* arow = ctxh + (size_t)(rowbase + l16) * CC;
  const _Float16* brow[8];
#pragma unroll
  for (int dc = 0; dc < 8; ++dc)
    brow[dc] = woh + (size_t)(colbase + dc * 16 + l16) * CC;

  v8f acc[8] = {};
  for (int kb = 0; kb < CC; kb += 32) {
    v16h a = load_a_frag(arow, kb, half);
    __builtin_prefetch((const void*)(arow + kb + 32), 0, 3);
#pragma unroll
    for (int dc = 0; dc < 8; ++dc) {
      v16h bf = load_b_frag(brow[dc], kb, half);
      acc[dc] = wmma_f16(a, bf, acc[dc]);
    }
  }
#pragma unroll
  for (int dc = 0; dc < 8; ++dc) {
#pragma unroll
    for (int r = 0; r < 8; ++r) {
      int m = rowbase + r + 8 * half;
      int n = colbase + dc * 16 + l16;
      xout[(size_t)m * CC + n] = acc[dc][r];
    }
  }
}

// conv1d(k=3, zero pad) + mean/std(unbiased) normalize + softmax(T=0.1)
__global__ void __launch_bounds__(256)
final_kernel(const float* __restrict__ imp, const float* __restrict__ smw,
             float* __restrict__ out) {
  __shared__ float s[NN];
  __shared__ float red[32];
  const int b = blockIdx.x;
  const int tid = threadIdx.x;
  const float w0 = smw[0], w1 = smw[1], w2 = smw[2];
  const float* ip = imp + (size_t)b * NN;

  for (int i = tid; i < NN; i += blockDim.x) {
    float l = (i > 0) ? ip[i - 1] : 0.f;
    float m = ip[i];
    float r = (i < NN - 1) ? ip[i + 1] : 0.f;
    s[i] = w0 * l + w1 * m + w2 * r;
  }
  __syncthreads();

  float v = 0.f;
  for (int i = tid; i < NN; i += blockDim.x) v += s[i];
  float mu = block_sum(v, red) / (float)NN;

  float v2 = 0.f;
  for (int i = tid; i < NN; i += blockDim.x) {
    float d = s[i] - mu;
    v2 += d * d;
  }
  float var = block_sum(v2, red) / (float)(NN - 1);
  float inv = 1.0f / (sqrtf(var) + 1e-6f);

  float mx = -3.4e38f;
  for (int i = tid; i < NN; i += blockDim.x) {
    float z = (s[i] - mu) * inv * 10.0f;  // /TEMP, TEMP=0.1
    s[i] = z;
    mx = fmaxf(mx, z);
  }
  mx = block_max(mx, red);

  float se = 0.f;
  for (int i = tid; i < NN; i += blockDim.x) {
    float e = __expf(s[i] - mx);
    s[i] = e;
    se += e;
  }
  float tot = block_sum(se, red);
  float itot = 1.0f / tot;
  for (int i = tid; i < NN; i += blockDim.x)
    out[(size_t)b * NN + i] = s[i] * itot;
}

// ---------------------------------------------------------------- launch

extern "C" void kernel_launch(void* const* d_in, const int* in_sizes, int n_in,
                              void* d_out, int out_size, void* d_ws,
                              size_t ws_size, hipStream_t stream) {
  (void)in_sizes; (void)n_in; (void)out_size; (void)ws_size;
  const float* x = (const float*)d_in[0];      // [B,N,C]
  const float* w_in = (const float*)d_in[1];   // [3C,C]
  const float* w_out = (const float*)d_in[2];  // [C,C]
  const float* sm_w = (const float*)d_in[3];   // [3]

  float* xout = (float*)d_out;                   // B*N*C
  float* fscores = xout + (size_t)BB * NN * CC;  // B*N

  // workspace carve-up (all sizes 256B-aligned)
  char* ws = (char*)d_ws;
  size_t off = 0;
  auto carve = [&](size_t bytes) {
    void* p = ws + off;
    off += (bytes + 255) & ~(size_t)255;
    return p;
  };
  _Float16* xh  = (_Float16*)carve((size_t)BB * NN * CC * 2);
  _Float16* wih = (_Float16*)carve((size_t)TC * CC * 2);
  _Float16* woh = (_Float16*)carve((size_t)CC * CC * 2);
  _Float16* qh  = (_Float16*)carve((size_t)BB * HH * NN * HD * 2);
  _Float16* kh  = (_Float16*)carve((size_t)BB * HH * NN * HD * 2);
  _Float16* vT  = (_Float16*)carve((size_t)BB * HH * HD * NN * 2);
  _Float16* ctxh = (_Float16*)carve((size_t)BB * NN * CC * 2);
  float* aw  = (float*)carve((size_t)BB * NN * NN * 4);
  float* imp = (float*)carve((size_t)BB * NN * 4);

  const int nX = BB * NN * CC;
  const int nWi = TC * CC;
  const int nWo = CC * CC;
  const int nAw = BB * NN * NN;

  cvt_f32_to_f16<<<(nX + 255) / 256, 256, 0, stream>>>(x, xh, nX);
  cvt_f32_to_f16<<<(nWi + 255) / 256, 256, 0, stream>>>(w_in, wih, nWi);
  cvt_f32_to_f16<<<(nWo + 255) / 256, 256, 0, stream>>>(w_out, woh, nWo);
  zero_f32<<<(nAw + 255) / 256, 256, 0, stream>>>(aw, nAw);

  dim3 gQkv((BB * NN) / 16, TC / 128);
  qkv_gemm<<<gQkv, 32, 0, stream>>>(xh, wih, qh, kh, vT);

  dim3 gAttn(NN / 16, HH, BB);
  attn_kernel<<<gAttn, 32, 0, stream>>>(qh, kh, vT, ctxh, aw);

  grad_kernel<<<(BB * NN + 255) / 256, 256, 0, stream>>>(aw, imp);

  dim3 gOut((BB * NN) / 16, CC / 128);
  outproj_gemm<<<gOut, 32, 0, stream>>>(ctxh, woh, xout);

  final_kernel<<<BB, 256, 0, stream>>>(imp, sm_w, fscores);
}
